// vaeAttn_3126736191840
// MI455X (gfx1250) — matrix operand order repaired and checked by reading, not compiled, not run on hardware
//
#include <hip/hip_runtime.h>

// ---------------- problem constants (match reference) ----------------
#define NHEADS   16
#define HDIM     64
#define EMB_SZ   1024
#define DMODEL   1024
#define NQKV     3072
#define BATCH    2
#define SEQ      2048
#define MROWS    (BATCH * SEQ)          // 4096 flattened rows

// ---------------- WMMA types (CDNA5 gfx1250, wave32) ----------------
typedef __attribute__((ext_vector_type(16))) __bf16 bf16x16;   // A/B operand: 8 VGPRs
typedef __attribute__((ext_vector_type(8)))  float  floatx8;   // C/D operand: 8 VGPRs

union FragBF { bf16x16 v; uint4 q[2]; };   // two 16B halves per fragment

__device__ __forceinline__ unsigned short f2bf(float f) {
  // round-to-nearest-even fp32 -> bf16
  unsigned u = __float_as_uint(f);
  u += 0x7FFFu + ((u >> 16) & 1u);
  return (unsigned short)(u >> 16);
}

__device__ __forceinline__ floatx8 wmma_bf16(const bf16x16& a, const bf16x16& b, floatx8 c) {
  return __builtin_amdgcn_wmma_f32_16x16x32_bf16(false, a, false, b, (short)0, c,
                                                 false, false);
}

// A-operand (ISA A-layout, 16x32): lane<16 holds K={kb+0..7, kb+16..23},
// lane>=16 holds K={kb+8..15, kb+24..31}; row = lane&15 baked into rowp.
__device__ __forceinline__ void loadA(FragBF& a, const unsigned short* rowp, int kb, int hi) {
  const unsigned short* ap = rowp + kb + hi * 8;
  a.q[0] = *(const uint4*)(ap);
  a.q[1] = *(const uint4*)(ap + 16);
}
// B-operand (32x16): lane%16 = N (baked into colp), K order same as A:
// lane<16 holds K={kb+0..7, kb+16..23}, lane>=16 holds K={kb+8..15, kb+24..31}.
__device__ __forceinline__ void loadB(FragBF& b, const unsigned short* colp, int kb, int hi) {
  const unsigned short* bp = colp + kb + hi * 8;
  b.q[0] = *(const uint4*)(bp);
  b.q[1] = *(const uint4*)(bp + 16);
}

// ---------------- conversion kernels ----------------
__global__ void cvt_f32_bf16_vec4(const float* __restrict__ in,
                                  unsigned short* __restrict__ out, int n4) {
  int i = blockIdx.x * blockDim.x + threadIdx.x;
  if (i >= n4) return;
  float4 f = ((const float4*)in)[i];
  ushort4 o;
  o.x = f2bf(f.x); o.y = f2bf(f.y); o.z = f2bf(f.z); o.w = f2bf(f.w);
  ((ushort4*)out)[i] = o;
}

// W [K][N] fp32 -> Wt [N][K] bf16 so B-operand reads are contiguous in K
__global__ void cvt_transpose_bf16(const float* __restrict__ in,
                                   unsigned short* __restrict__ out, int K, int N) {
  int i = blockIdx.x * blockDim.x + threadIdx.x;
  if (i >= K * N) return;
  int k = i / N, n = i - k * N;
  out[n * K + k] = f2bf(in[i]);
}

// ---------------- pipelined GEMM wave core: 32M x 64N per wave ----------------
__device__ __forceinline__ void gemm_load_step(FragBF a[2], FragBF b[4],
                                               const unsigned short* ar0,
                                               const unsigned short* ar1,
                                               const unsigned short* const (&bc)[4],
                                               int kb, int hi) {
  loadA(a[0], ar0, kb, hi);
  loadA(a[1], ar1, kb, hi);
#pragma unroll
  for (int t = 0; t < 4; ++t) loadB(b[t], bc[t], kb, hi);
}

__device__ __forceinline__ void gemm_mma_step(FragBF a[2], FragBF b[4], floatx8 acc[2][4]) {
#pragma unroll
  for (int mt = 0; mt < 2; ++mt)
#pragma unroll
    for (int t = 0; t < 4; ++t)
      acc[mt][t] = wmma_bf16(a[mt].v, b[t].v, acc[mt][t]);
}

// Double-buffered K loop: next step's fragments are in flight while the
// current step's 8 WMMAs execute (avoids s_wait_loadcnt 0 before every WMMA).
template <int K>
__device__ __forceinline__ void gemm_core(const unsigned short* ar0,
                                          const unsigned short* ar1,
                                          const unsigned short* const (&bc)[4],
                                          int hi, floatx8 acc[2][4]) {
  FragBF a0[2], b0[4], a1[2], b1[4];
  gemm_load_step(a0, b0, ar0, ar1, bc, 0, hi);
  for (int kb = 0; kb + 64 <= K; kb += 64) {
    gemm_load_step(a1, b1, ar0, ar1, bc, kb + 32, hi);
    gemm_mma_step(a0, b0, acc);
    if (kb + 64 < K) gemm_load_step(a0, b0, ar0, ar1, bc, kb + 64, hi);
    gemm_mma_step(a1, b1, acc);
  }
}

// ---------------- QKV projection GEMM ----------------
// C[4096,3072] = Xbf[4096,1024] @ Wqkv; epilogue adds bias and scatters into
// Q[b,h,l,d], K[b,h,l,d], Vt[b,h,d,l] (bf16). One wave owns a 32x64 tile.
__global__ __launch_bounds__(128)
void qkv_gemm(const unsigned short* __restrict__ xbf,   // [4096][1024]
              const unsigned short* __restrict__ wt,    // [3072][1024] (n-major)
              const float* __restrict__ bias,           // [3072]
              unsigned short* __restrict__ qws,
              unsigned short* __restrict__ kws,
              unsigned short* __restrict__ vwsT) {
  const int lane = threadIdx.x;
  const int wave = threadIdx.y;
  const int m0 = (blockIdx.y * 4 + wave) * 32;
  const int n0 = blockIdx.x * 64;
  const int mlo = lane & 15;
  const int hi  = lane >> 4;

  const unsigned short* ar0 = xbf + (m0 + mlo) * EMB_SZ;
  const unsigned short* ar1 = xbf + (m0 + 16 + mlo) * EMB_SZ;
  const unsigned short* bc[4];
#pragma unroll
  for (int t = 0; t < 4; ++t) bc[t] = wt + (n0 + t * 16 + mlo) * EMB_SZ;

  floatx8 acc[2][4] = {};
  gemm_core<EMB_SZ>(ar0, ar1, bc, hi, acc);

  // epilogue: C-layout row = r + 8*hi, col = mlo within tile
#pragma unroll
  for (int mt = 0; mt < 2; ++mt) {
#pragma unroll
    for (int t = 0; t < 4; ++t) {
      int n = n0 + t * 16 + mlo;
      float bb = bias[n];
      int which = n >> 10;               // 0=q 1=k 2=v
      int h = (n & 1023) >> 6;
      int d = n & 63;
#pragma unroll
      for (int r = 0; r < 8; ++r) {
        int m = m0 + mt * 16 + r + 8 * hi;   // global row = b*SEQ + l
        int b = m >> 11, l = m & (SEQ - 1);
        unsigned short u = f2bf(acc[mt][t][r] + bb);
        if (which == 0)      qws[((b * NHEADS + h) * SEQ + l) * HDIM + d] = u;
        else if (which == 1) kws[((b * NHEADS + h) * SEQ + l) * HDIM + d] = u;
        else                 vwsT[((b * NHEADS + h) * HDIM + d) * SEQ + l] = u;
      }
    }
  }
}

// ---------------- fused flash attention ----------------
// One wave per (b, h, 32-row q tile): 2 q-tiles share every K/V chunk.
// Online softmax over 32-wide KV chunks; 16 WMMAs per chunk.
__global__ __launch_bounds__(32)
void flash_attn(const unsigned short* __restrict__ qws,   // [b][h][l][d]
                const unsigned short* __restrict__ kws,   // [b][h][l][d]
                const unsigned short* __restrict__ vwsT,  // [b][h][d][l]
                unsigned short* __restrict__ attn) {      // [b*SEQ + l][h*64 + d]
  __shared__ __align__(16) unsigned short Pld[32 * 32];   // P tile, C->A relayout

  const int lane = threadIdx.x;
  const int tile = blockIdx.x;                 // b*H*64 + h*64 + qb
  const int qb = tile & 63;                    // 32-row q block
  const int h  = (tile >> 6) & (NHEADS - 1);
  const int b  = tile >> 10;
  const unsigned short* Qb = qws  + ((b * NHEADS + h) * SEQ) * HDIM;
  const unsigned short* Kb = kws  + ((b * NHEADS + h) * SEQ) * HDIM;
  const unsigned short* Vb = vwsT + ((b * NHEADS + h) * HDIM) * SEQ;

  const int mlo = lane & 15;
  const int hi  = lane >> 4;

  // Q: 2 q-tiles x 2 d-slices as A fragments, register-resident all kernel
  FragBF qa[2][2];
#pragma unroll
  for (int qt = 0; qt < 2; ++qt) {
    const unsigned short* qp = Qb + (qb * 32 + qt * 16 + mlo) * HDIM;
    loadA(qa[qt][0], qp, 0, hi);
    loadA(qa[qt][1], qp, 32, hi);
  }

  floatx8 O[2][4] = {};              // 32x64 f32 accumulator
  float mrow[2][8], lrow[2][8];
#pragma unroll
  for (int qt = 0; qt < 2; ++qt)
#pragma unroll
    for (int r = 0; r < 8; ++r) { mrow[qt][r] = -3.0e38f; lrow[qt][r] = 0.0f; }
  const float sc = 0.125f;           // 1/sqrt(64)

  for (int kv = 0; kv < SEQ; kv += 32) {
    // ---- issue ALL K and V fragment loads for this chunk up front ----
    FragBF kf[2][2];                 // [n-tile][d-slice]
#pragma unroll
    for (int nt = 0; nt < 2; ++nt) {
      const unsigned short* kp = Kb + (kv + nt * 16 + mlo) * HDIM;
      loadB(kf[nt][0], kp, 0, hi);
      loadB(kf[nt][1], kp, 32, hi);
    }
    FragBF vf[4];                    // V^T: B-operand, contiguous in kv
#pragma unroll
    for (int dt = 0; dt < 4; ++dt)
      loadB(vf[dt], Vb + (dt * 16 + mlo) * SEQ, kv, hi);
    if (kv + 32 < SEQ)
      __builtin_prefetch(Kb + (kv + 32 + mlo) * HDIM, 0, 0);

    // ---- scores S[32 x 32] = Q K^T : 8 WMMAs (V loads still in flight) ----
    floatx8 s[2][2] = {};
#pragma unroll
    for (int qt = 0; qt < 2; ++qt)
#pragma unroll
      for (int nt = 0; nt < 2; ++nt) {
        s[qt][nt] = wmma_bf16(qa[qt][0].v, kf[nt][0].v, s[qt][nt]);
        s[qt][nt] = wmma_bf16(qa[qt][1].v, kf[nt][1].v, s[qt][nt]);
      }

    __syncthreads();   // previous chunk's P loads done before overwrite

    // ---- online softmax; rows live across 16 lanes of one half ----
#pragma unroll
    for (int qt = 0; qt < 2; ++qt) {
#pragma unroll
      for (int r = 0; r < 8; ++r) {
        float v0 = s[qt][0][r] * sc, v1 = s[qt][1][r] * sc;
        float mx = fmaxf(v0, v1);
        mx = fmaxf(mx, __shfl_xor(mx, 1, 32));
        mx = fmaxf(mx, __shfl_xor(mx, 2, 32));
        mx = fmaxf(mx, __shfl_xor(mx, 4, 32));
        mx = fmaxf(mx, __shfl_xor(mx, 8, 32));
        float mnew  = fmaxf(mrow[qt][r], mx);
        float alpha = __expf(mrow[qt][r] - mnew);
        float p0 = __expf(v0 - mnew);
        float p1 = __expf(v1 - mnew);
        float ps = p0 + p1;
        ps += __shfl_xor(ps, 1, 32);
        ps += __shfl_xor(ps, 2, 32);
        ps += __shfl_xor(ps, 4, 32);
        ps += __shfl_xor(ps, 8, 32);
        lrow[qt][r] = lrow[qt][r] * alpha + ps;
        mrow[qt][r] = mnew;
#pragma unroll
        for (int t = 0; t < 4; ++t) O[qt][t][r] *= alpha;
        int row = qt * 16 + r + 8 * hi;
        Pld[row * 32 + mlo]      = f2bf(p0);
        Pld[row * 32 + 16 + mlo] = f2bf(p1);
      }
    }
    __syncthreads();

    // ---- O += P V : reload P in A-layout; V^T fragments already resident ----
    FragBF pa[2];
#pragma unroll
    for (int qt = 0; qt < 2; ++qt)
      loadA(pa[qt], Pld + (qt * 16 + mlo) * 32, 0, hi);
#pragma unroll
    for (int qt = 0; qt < 2; ++qt)
#pragma unroll
      for (int dt = 0; dt < 4; ++dt)
        O[qt][dt] = wmma_bf16(pa[qt].v, vf[dt].v, O[qt][dt]);
  }

  // ---- finalize: O/l, write bf16 [b*SEQ + l][h*64 + d] ----
#pragma unroll
  for (int qt = 0; qt < 2; ++qt) {
#pragma unroll
    for (int r = 0; r < 8; ++r) {
      float inv = 1.0f / lrow[qt][r];
      int row = b * SEQ + qb * 32 + qt * 16 + r + 8 * hi;
#pragma unroll
      for (int dt = 0; dt < 4; ++dt) {
        int col = h * HDIM + dt * 16 + mlo;
        attn[row * DMODEL + col] = f2bf(O[qt][dt][r] * inv);
      }
    }
  }
}

// ---------------- output projection GEMM ----------------
// d_out[4096,1024] = attn_bf16 @ W_out + b_out  (f32 epilogue)
__global__ __launch_bounds__(128)
void out_gemm(const unsigned short* __restrict__ abf,  // [4096][1024]
              const unsigned short* __restrict__ wt,   // [1024][1024] n-major
              const float* __restrict__ bias,          // [1024]
              float* __restrict__ out) {
  const int lane = threadIdx.x;
  const int wave = threadIdx.y;
  const int m0 = (blockIdx.y * 4 + wave) * 32;
  const int n0 = blockIdx.x * 64;
  const int mlo = lane & 15;
  const int hi  = lane >> 4;

  const unsigned short* ar0 = abf + (m0 + mlo) * DMODEL;
  const unsigned short* ar1 = abf + (m0 + 16 + mlo) * DMODEL;
  const unsigned short* bc[4];
#pragma unroll
  for (int t = 0; t < 4; ++t) bc[t] = wt + (n0 + t * 16 + mlo) * DMODEL;

  floatx8 acc[2][4] = {};
  gemm_core<DMODEL>(ar0, ar1, bc, hi, acc);

#pragma unroll
  for (int mt = 0; mt < 2; ++mt) {
#pragma unroll
    for (int t = 0; t < 4; ++t) {
      int n = n0 + t * 16 + mlo;
      float bb = bias[n];
#pragma unroll
      for (int r = 0; r < 8; ++r) {
        int m = m0 + mt * 16 + r + 8 * hi;
        out[m * EMB_SZ + n] = acc[mt][t][r] + bb;
      }
    }
  }
}

// ---------------- host launch ----------------
extern "C" void kernel_launch(void* const* d_in, const int* in_sizes, int n_in,
                              void* d_out, int out_size, void* d_ws, size_t ws_size,
                              hipStream_t stream) {
  const float* x     = (const float*)d_in[0];   // [2,2048,1024]
  const float* W_qkv = (const float*)d_in[1];   // [1024,3072]
  const float* b_qkv = (const float*)d_in[2];   // [3072]
  const float* W_out = (const float*)d_in[3];   // [1024,1024]
  const float* b_out = (const float*)d_in[4];   // [1024]
  float* out = (float*)d_out;

  char* ws = (char*)d_ws;
  size_t off = 0;
  auto take = [&](size_t bytes) { char* p = ws + off; off += (bytes + 255) & ~size_t(255); return p; };
  unsigned short* xbf   = (unsigned short*)take((size_t)MROWS * EMB_SZ * 2);   // 8 MB
  unsigned short* wqkvT = (unsigned short*)take((size_t)NQKV * EMB_SZ * 2);    // 6 MB
  unsigned short* woutT = (unsigned short*)take((size_t)DMODEL * EMB_SZ * 2);  // 2 MB
  unsigned short* qws   = (unsigned short*)take((size_t)MROWS * DMODEL * 2);   // 8 MB
  unsigned short* kws   = (unsigned short*)take((size_t)MROWS * DMODEL * 2);   // 8 MB
  unsigned short* vwsT  = (unsigned short*)take((size_t)MROWS * DMODEL * 2);   // 8 MB
  unsigned short* attn  = (unsigned short*)take((size_t)MROWS * DMODEL * 2);   // 8 MB

  // 1) precision conversion (+ weight transpose for WMMA B-operand layout)
  {
    int n4 = MROWS * EMB_SZ / 4;
    cvt_f32_bf16_vec4<<<n4 / 256, 256, 0, stream>>>(x, xbf, n4);
  }
  cvt_transpose_bf16<<<(EMB_SZ * NQKV + 255) / 256, 256, 0, stream>>>(W_qkv, wqkvT, EMB_SZ, NQKV);
  cvt_transpose_bf16<<<(DMODEL * EMB_SZ + 255) / 256, 256, 0, stream>>>(W_out, woutT, DMODEL, EMB_SZ);

  // 2) QKV projection -> Q/K row-major, V transposed (32x64 tile per wave)
  qkv_gemm<<<dim3(NQKV / 64, MROWS / 128), dim3(32, 4), 0, stream>>>(
      xbf, wqkvT, b_qkv, qws, kws, vwsT);

  // 3) fused flash attention: one wave per (b, h, 32-row q tile)
  flash_attn<<<BATCH * NHEADS * (SEQ / 32), 32, 0, stream>>>(qws, kws, vwsT, attn);

  // 4) output projection (f32 + bias epilogue straight to d_out)
  out_gemm<<<dim3(EMB_SZ / 64, MROWS / 128), dim3(32, 4), 0, stream>>>(
      attn, woutT, b_out, out);

  (void)in_sizes; (void)n_in; (void)out_size; (void)ws_size;
}
